// AutoRegressiveTransformer_75634374083398
// MI455X (gfx1250) — matrix operand order repaired and checked by reading, not compiled, not run on hardware
//
#include <hip/hip_runtime.h>

// ---------------------------------------------------------------------------
// AutoRegressive transformer (non-causal re-decode each step) for MI455X.
// All GEMMs via v_wmma_f32_16x16x32_f16 (f16 in, f32 accumulate).
// Weights converted once per call to f16, K-major (transposed) in d_ws.
// ---------------------------------------------------------------------------

typedef _Float16 h16;
typedef __attribute__((ext_vector_type(16))) _Float16 v16h;
typedef __attribute__((ext_vector_type(8)))  float    v8f;
typedef __attribute__((ext_vector_type(4)))  float    v4f;

#define DM    64
#define NH    4
#define DH    16
#define DFF   2048
#define NB    8
#define HLEN  25
#define FLEN  275
#define NPAD  288                       // 276 valid positions rounded to 18 tiles
#define BSTR  ((size_t)NPAD * DM)       // stream buffer batch stride (halves)
#define QSTR  ((size_t)NPAD * 192)     // qkv buffer batch stride
#define CSTR  ((size_t)32 * 192)       // cross-kv buffer batch stride (25->32 rows)

// ------------------------------- WMMA helpers ------------------------------

// A fragment: 16x32 f16 tile, row-major source with leading dim lda.
// lane l (l<16): M=l, K = {0..7, 16..23}; lane l>=16: M=l-16, K = {8..15, 24..31}.
__device__ __forceinline__ v16h load_a16(const h16* A, int lda) {
  const int l = threadIdx.x & 31;
  const h16* p = A + (size_t)(l & 15) * lda + ((l >> 4) << 3);
  v16h a;
  reinterpret_cast<v4f*>(&a)[0] = *reinterpret_cast<const v4f*>(p);
  reinterpret_cast<v4f*>(&a)[1] = *reinterpret_cast<const v4f*>(p + 16);
  return a;
}

// B fragment: 32x16 f16 tile from K-major weights (ldb = full N); lane l holds
// column N=l&15, K = {0..7, 16..23} (l<16) or {8..15, 24..31} (l>=16).
__device__ __forceinline__ v16h load_b16(const h16* Bt, int ldb) {
  const int l = threadIdx.x & 31;
  const h16* p = Bt + (l & 15) + (size_t)((l >> 4) << 3) * ldb;
  v16h b;
#pragma unroll
  for (int s = 0; s < 16; ++s) b[s] = p[(size_t)(s < 8 ? s : s + 8) * ldb];
  return b;
}

__device__ __forceinline__ v8f wmma_acc(const h16* A, int lda,
                                        const h16* Bt, int ldb, int K, v8f c) {
  for (int k = 0; k < K; k += 32) {
    v16h a = load_a16(A + k, lda);
    v16h b = load_b16(Bt + (size_t)k * ldb, ldb);
    c = __builtin_amdgcn_wmma_f32_16x16x32_f16(false, a, false, b, (short)0, c,
                                               false, false);
  }
  return c;
}

// C/D layout: lane l, vgpr r -> element (M = r + 8*(l>>4), N = l&15)
__device__ __forceinline__ void store_c_f16_bias(v8f c, const float* bias,
                                                 h16* out, int ldo, bool relu) {
  const int l = threadIdx.x & 31;
  const int n = l & 15, hi = l >> 4;
  const float bv = bias[n];
#pragma unroll
  for (int r = 0; r < 8; ++r) {
    float v = c[r] + bv;
    if (relu) v = fmaxf(v, 0.f);
    out[(size_t)(r + 8 * hi) * ldo + n] = (h16)v;
  }
}

__device__ __forceinline__ void store_c_f32(v8f c, float* out, int ldo) {
  const int l = threadIdx.x & 31;
  const int n = l & 15, hi = l >> 4;
#pragma unroll
  for (int r = 0; r < 8; ++r) out[(size_t)(r + 8 * hi) * ldo + n] = c[r];
}

__device__ __forceinline__ float wred_sum(float v) {
#pragma unroll
  for (int o = 16; o >= 1; o >>= 1) v += __shfl_xor(v, o, 32);
  return v;
}
__device__ __forceinline__ float wred_max(float v) {
#pragma unroll
  for (int o = 16; o >= 1; o >>= 1) v = fmaxf(v, __shfl_xor(v, o, 32));
  return v;
}

// ------------------------------- kernels -----------------------------------

// Generic f16 GEMM with f32 bias, f16 output: C[b, mt*16.., n0..] = A @ Bt + bias
__global__ __launch_bounds__(32) void gemm_bias_f16_kernel(
    const h16* __restrict__ A, size_t a_bstr, int lda,
    const h16* __restrict__ Bt, int ldb, const float* __restrict__ bias,
    h16* __restrict__ C, size_t c_bstr, int ldc, int Ntiles, int K) {
  const int b = blockIdx.y;
  const int mt = blockIdx.x / Ntiles;
  const int nt = blockIdx.x - mt * Ntiles;
  const int n0 = nt * 16;
  v8f c = {};
  c = wmma_acc(A + (size_t)b * a_bstr + (size_t)mt * 16 * lda, lda,
               Bt + n0, ldb, K, c);
  store_c_f16_bias(c, bias + n0,
                   C + (size_t)b * c_bstr + (size_t)mt * 16 * ldc + n0, ldc, false);
}

// Softmax attention: one wave per (query, batch); loops heads. dh=16, scale 1/4.
// Q rows: [.. j*192 + 0..63], K rows: [.. j*192 + 64..], V rows: [.. +128..].
__global__ __launch_bounds__(32) void attn_kernel(
    const h16* __restrict__ Q, size_t q_bstr,
    const h16* __restrict__ KV, size_t kv_bstr,
    h16* __restrict__ O, size_t o_bstr, int nkeys) {
  const int jq = blockIdx.x, b = blockIdx.y, lane = threadIdx.x;
  const h16* qrow = Q + (size_t)b * q_bstr + (size_t)jq * 192;
  const h16* kvb = KV + (size_t)b * kv_bstr;
  h16* orow = O + (size_t)b * o_bstr + (size_t)jq * DM;
  for (int h = 0; h < NH; ++h) {
    float q[DH];
#pragma unroll
    for (int i = 0; i < DH; ++i) q[i] = (float)qrow[h * DH + i];
    float s[9];
    float mx = -1e30f;
    int cnt = 0;
    for (int j = lane; j < nkeys; j += 32, ++cnt) {
      const h16* kr = kvb + (size_t)j * 192 + 64 + h * DH;
      float d = 0.f;
#pragma unroll
      for (int i = 0; i < DH; ++i) d += q[i] * (float)kr[i];
      d *= 0.25f;  // 1/sqrt(16)
      s[cnt] = d;
      mx = fmaxf(mx, d);
    }
    mx = wred_max(mx);
    float sum = 0.f;
    for (int c2 = 0; c2 < cnt; ++c2) { s[c2] = __expf(s[c2] - mx); sum += s[c2]; }
    sum = wred_sum(sum);
    const float inv = 1.f / sum;
    float oa[DH];
#pragma unroll
    for (int i = 0; i < DH; ++i) oa[i] = 0.f;
    cnt = 0;
    for (int j = lane; j < nkeys; j += 32, ++cnt) {
      const h16* vr = kvb + (size_t)j * 192 + 128 + h * DH;
      const float a = s[cnt] * inv;
#pragma unroll
      for (int i = 0; i < DH; ++i) oa[i] += a * (float)vr[i];
    }
#pragma unroll
    for (int i = 0; i < DH; ++i) {
      float v = wred_sum(oa[i]);
      if (lane == 0) orow[h * DH + i] = (h16)v;
    }
  }
}

// out-proj (64x64) + bias + residual + LayerNorm, fused.  block = 4 waves.
__global__ __launch_bounds__(128) void oproj_ln_kernel(
    const h16* __restrict__ O, size_t o_bstr,
    const h16* __restrict__ Wot, const float* __restrict__ bo,
    const h16* __restrict__ Xres, size_t x_bstr,
    const float* __restrict__ lnw, const float* __restrict__ lnb,
    h16* __restrict__ Xout, size_t xo_bstr) {
  __shared__ float P[16 * DM];
  const int b = blockIdx.y, mt = blockIdx.x;
  const int wv = threadIdx.x >> 5, lane = threadIdx.x & 31;
  v8f c = {};
  c = wmma_acc(O + (size_t)b * o_bstr + (size_t)mt * 16 * DM, DM,
               Wot + wv * 16, DM, DM, c);
  store_c_f32(c, P + wv * 16, DM);
  __syncthreads();
  for (int r = wv; r < 16; r += 4) {
    const size_t row = (size_t)b * x_bstr + (size_t)(mt * 16 + r) * DM;
    const int d0 = lane, d1 = lane + 32;
    float v0 = (float)Xres[row + d0] + P[r * DM + d0] + bo[d0];
    float v1 = (float)Xres[row + d1] + P[r * DM + d1] + bo[d1];
    const float m = wred_sum(v0 + v1) * (1.f / 64.f);
    v0 -= m; v1 -= m;
    const float var = wred_sum(v0 * v0 + v1 * v1) * (1.f / 64.f);
    const float rs = rsqrtf(var + 1e-5f);
    const size_t orow = (size_t)b * xo_bstr + (size_t)(mt * 16 + r) * DM;
    Xout[orow + d0] = (h16)(v0 * rs * lnw[d0] + lnb[d0]);
    Xout[orow + d1] = (h16)(v1 * rs * lnw[d1] + lnb[d1]);
  }
}

// Fully fused FFN: y = LN(x + relu(x@W1t + b1)@W2t + b2).  Hidden tile kept in
// LDS f16, processed in two 1024-wide halves; split-K FFN2 across waves.
__global__ __launch_bounds__(256) void ffn_ln_kernel(
    const h16* __restrict__ X, size_t x_bstr,
    const h16* __restrict__ W1t, const float* __restrict__ b1,
    const h16* __restrict__ W2t, const float* __restrict__ b2,
    const float* __restrict__ lnw, const float* __restrict__ lnb,
    h16* __restrict__ Xout, size_t xo_bstr) {
  __shared__ __align__(32) h16 XT[16 * DM];
  __shared__ __align__(32) h16 HID[16 * 1024];
  __shared__ float P[2][16 * DM];
  const int b = blockIdx.y, mt = blockIdx.x;
  const int tid = threadIdx.x, wv = tid >> 5, lane = tid & 31;
  const h16* Xb = X + (size_t)b * x_bstr + (size_t)mt * 16 * DM;
  for (int i = tid; i < 16 * DM; i += 256) XT[i] = Xb[i];
  __syncthreads();
  const int nt = wv & 3, kh = wv >> 2;
  v8f acc = {};
  for (int hf = 0; hf < 2; ++hf) {
    for (int tt = wv; tt < 64; tt += 8) {          // FFN1: 64 n-tiles this half
      const int n0g = hf * 1024 + tt * 16;
      v8f c = {};
      c = wmma_acc(XT, DM, W1t + n0g, DFF, DM, c);
      store_c_f16_bias(c, b1 + n0g, HID + tt * 16, 1024, true);
    }
    __syncthreads();
    // FFN2 partial: wave (nt, kh) accumulates k in [kh*512, kh*512+512)
    acc = wmma_acc(HID + kh * 512, 1024,
                   W2t + (size_t)(hf * 1024 + kh * 512) * DM + nt * 16, DM,
                   512, acc);
    __syncthreads();                                // before HID is overwritten
  }
  store_c_f32(acc, P[kh] + nt * 16, DM);
  __syncthreads();
  for (int r = wv; r < 16; r += 8) {
    const int d0 = lane, d1 = lane + 32;
    float v0 = (float)XT[r * DM + d0] + P[0][r * DM + d0] + P[1][r * DM + d0] + b2[d0];
    float v1 = (float)XT[r * DM + d1] + P[0][r * DM + d1] + P[1][r * DM + d1] + b2[d1];
    const float m = wred_sum(v0 + v1) * (1.f / 64.f);
    v0 -= m; v1 -= m;
    const float var = wred_sum(v0 * v0 + v1 * v1) * (1.f / 64.f);
    const float rs = rsqrtf(var + 1e-5f);
    const size_t orow = (size_t)b * xo_bstr + (size_t)(mt * 16 + r) * DM;
    Xout[orow + d0] = (h16)(v0 * rs * lnw[d0] + lnb[d0]);
    Xout[orow + d1] = (h16)(v1 * rs * lnw[d1] + lnb[d1]);
  }
}

__global__ __launch_bounds__(64) void commit_kernel(
    const h16* __restrict__ A, size_t a_bstr,
    h16* __restrict__ bufX, size_t x_bstr, float* __restrict__ out, int t) {
  const int b = blockIdx.x, d = threadIdx.x;
  const float v = (float)A[(size_t)b * a_bstr + (size_t)t * DM + d];
  out[((size_t)b * FLEN + t) * DM + d] = v;
  bufX[(size_t)b * x_bstr + (size_t)(t + 1) * DM + d] = (h16)v;
}

__global__ __launch_bounds__(256) void transpose_w_kernel(
    const float* __restrict__ W, h16* __restrict__ Wt, int N, int K) {
  const int i = blockIdx.x * 256 + threadIdx.x;
  if (i >= N * K) return;
  const int n = i / K, k = i - n * K;
  Wt[(size_t)k * N + n] = (h16)W[(size_t)n * K + k];
}

__global__ __launch_bounds__(256) void cvt_hist_kernel(
    const float* __restrict__ hist, h16* __restrict__ Xe) {
  const int i = blockIdx.x * 256 + threadIdx.x;
  if (i >= NB * HLEN * DM) return;
  const int b = i / (HLEN * DM);
  const int rem = i - b * HLEN * DM;
  Xe[(size_t)b * BSTR + rem] = (h16)hist[i];
}

__global__ __launch_bounds__(256) void zero_h16_kernel(h16* p, int n) {
  const int i = blockIdx.x * 256 + threadIdx.x;
  if (i < n) p[i] = (h16)0.f;
}

// ------------------------------- host side ---------------------------------

struct LyrP {
  const float *saW, *sab, *saWo, *sabo, *W1, *b1, *W2, *b2;
  const float *ln1w, *ln1b, *ln2w, *ln2b;
  const float *caW, *cab, *caWo, *cabo, *ln3w, *ln3b;
};

extern "C" void kernel_launch(void* const* d_in, const int* in_sizes, int n_in,
                              void* d_out, int out_size, void* d_ws, size_t ws_size,
                              hipStream_t stream) {
  (void)in_sizes; (void)n_in; (void)out_size; (void)ws_size;
  const float* hist = (const float*)d_in[0];

  // Flattened input order: hist, enc[0..1] (12 tensors each, insertion order),
  // dec[0..1] (12 + 6 tensors each, insertion order).
  auto fp = [&](int i) { return (const float*)d_in[i]; };
  LyrP E[2], D[2];
  for (int l = 0; l < 2; ++l) {
    const int b0 = 1 + 12 * l;
    E[l] = {fp(b0+0), fp(b0+1), fp(b0+2), fp(b0+3), fp(b0+4), fp(b0+5),
            fp(b0+6), fp(b0+7), fp(b0+8), fp(b0+9), fp(b0+10), fp(b0+11),
            nullptr, nullptr, nullptr, nullptr, nullptr, nullptr};
  }
  for (int l = 0; l < 2; ++l) {
    const int b0 = 25 + 18 * l;
    D[l] = {fp(b0+0), fp(b0+1), fp(b0+2), fp(b0+3), fp(b0+4), fp(b0+5),
            fp(b0+6), fp(b0+7), fp(b0+8), fp(b0+9), fp(b0+10), fp(b0+11),
            fp(b0+12), fp(b0+13), fp(b0+14), fp(b0+15), fp(b0+16), fp(b0+17)};
  }

  // Workspace carve-up (f16 halves), ~5.2 MB total.
  h16* w = (h16*)d_ws;
  size_t off = 0;
  auto alloc = [&](size_t n) { h16* p = w + off; off += (n + 63) & ~(size_t)63; return p; };
  h16 *saWtE[2], *saWotE[2], *W1tE[2], *W2tE[2];
  h16 *saWtD[2], *saWotD[2], *W1tD[2], *W2tD[2], *caWtD[2], *caWotD[2];
  for (int l = 0; l < 2; ++l) {
    saWtE[l] = alloc(192 * 64); saWotE[l] = alloc(64 * 64);
    W1tE[l] = alloc(64 * 2048); W2tE[l] = alloc(2048 * 64);
  }
  for (int l = 0; l < 2; ++l) {
    saWtD[l] = alloc(192 * 64); saWotD[l] = alloc(64 * 64);
    W1tD[l] = alloc(64 * 2048); W2tD[l] = alloc(2048 * 64);
    caWtD[l] = alloc(192 * 64); caWotD[l] = alloc(64 * 64);
  }
  h16* SA  = alloc(NB * BSTR);
  h16* SB  = alloc(NB * BSTR);
  h16* SC  = alloc(NB * BSTR);
  h16* SD  = alloc(NB * BSTR);
  h16* BUFX = alloc(NB * BSTR);
  h16* OB  = alloc(NB * BSTR);
  h16* QKV = alloc(NB * QSTR);
  h16* CQ[2] = {alloc(NB * CSTR), alloc(NB * CSTR)};

  // ---- weight prep: f32 [N][K] -> f16 K-major [K][N] ----
  auto tr = [&](const float* W, h16* Wt, int N, int K) {
    const int n = N * K;
    transpose_w_kernel<<<(n + 255) / 256, 256, 0, stream>>>(W, Wt, N, K);
  };
  for (int l = 0; l < 2; ++l) {
    tr(E[l].saW, saWtE[l], 192, 64);  tr(E[l].saWo, saWotE[l], 64, 64);
    tr(E[l].W1, W1tE[l], 2048, 64);   tr(E[l].W2, W2tE[l], 64, 2048);
  }
  for (int l = 0; l < 2; ++l) {
    tr(D[l].saW, saWtD[l], 192, 64);  tr(D[l].saWo, saWotD[l], 64, 64);
    tr(D[l].W1, W1tD[l], 2048, 64);   tr(D[l].W2, W2tD[l], 64, 2048);
    tr(D[l].caW, caWtD[l], 192, 64);  tr(D[l].caWo, caWotD[l], 64, 64);
  }
  zero_h16_kernel<<<(NB * (int)BSTR + 255) / 256, 256, 0, stream>>>(BUFX, NB * (int)BSTR);
  cvt_hist_kernel<<<(NB * HLEN * DM + 255) / 256, 256, 0, stream>>>(hist, SA);

  auto sa_block = [&](const h16* Xin, const h16* Wt, const float* bb,
                      const h16* Wot, const float* bo,
                      const float* lnw, const float* lnb, h16* Xout, int ntok) {
    const int Mt = (ntok + 15) / 16;
    gemm_bias_f16_kernel<<<dim3(Mt * 12, NB), 32, 0, stream>>>(
        Xin, BSTR, DM, Wt, 192, bb, QKV, QSTR, 192, 12, DM);
    attn_kernel<<<dim3(ntok, NB), 32, 0, stream>>>(QKV, QSTR, QKV, QSTR, OB, BSTR, ntok);
    oproj_ln_kernel<<<dim3(Mt, NB), 128, 0, stream>>>(
        OB, BSTR, Wot, bo, Xin, BSTR, lnw, lnb, Xout, BSTR);
  };
  auto ca_block = [&](const h16* Xin, int li, h16* Xout, int ntok) {
    const int Mt = (ntok + 15) / 16;
    gemm_bias_f16_kernel<<<dim3(Mt * 4, NB), 32, 0, stream>>>(   // q only (cols 0..63)
        Xin, BSTR, DM, caWtD[li], 192, D[li].cab, QKV, QSTR, 192, 4, DM);
    attn_kernel<<<dim3(ntok, NB), 32, 0, stream>>>(QKV, QSTR, CQ[li], CSTR, OB, BSTR, HLEN);
    oproj_ln_kernel<<<dim3(Mt, NB), 128, 0, stream>>>(
        OB, BSTR, caWotD[li], D[li].cabo, Xin, BSTR, D[li].ln2w, D[li].ln2b, Xout, BSTR);
  };
  auto ffn_block = [&](const h16* Xin, const h16* W1t, const float* b1,
                       const h16* W2t, const float* b2,
                       const float* lnw, const float* lnb, h16* Xout, int ntok) {
    const int Mt = (ntok + 15) / 16;
    ffn_ln_kernel<<<dim3(Mt, NB), 256, 0, stream>>>(
        Xin, BSTR, W1t, b1, W2t, b2, lnw, lnb, Xout, BSTR);
  };

  // ---- encoder (2 layers, 25 tokens/batch) ----
  sa_block(SA, saWtE[0], E[0].sab, saWotE[0], E[0].sabo, E[0].ln1w, E[0].ln1b, SB, HLEN);
  ffn_block(SB, W1tE[0], E[0].b1, W2tE[0], E[0].b2, E[0].ln2w, E[0].ln2b, SC, HLEN);
  sa_block(SC, saWtE[1], E[1].sab, saWotE[1], E[1].sabo, E[1].ln1w, E[1].ln1b, SA, HLEN);
  ffn_block(SA, W1tE[1], E[1].b1, W2tE[1], E[1].b2, E[1].ln2w, E[1].ln2b, SB, HLEN);
  // mem = SB; precompute cross-attention K/V per decoder layer (mem is fixed).
  for (int l = 0; l < 2; ++l)
    gemm_bias_f16_kernel<<<dim3(2 * 12, NB), 32, 0, stream>>>(
        SB, BSTR, DM, caWtD[l], 192, D[l].cab, CQ[l], CSTR, 192, 12, DM);

  // ---- autoregressive decode: full (non-causal) re-decode over t+1 tokens ----
  for (int t = 0; t < FLEN; ++t) {
    const int ntok = t + 1;
    // layer 0: input = canonical buffer
    sa_block(BUFX, saWtD[0], D[0].sab, saWotD[0], D[0].sabo, D[0].ln1w, D[0].ln1b, SA, ntok);
    ca_block(SA, 0, SB, ntok);
    ffn_block(SB, W1tD[0], D[0].b1, W2tD[0], D[0].b2, D[0].ln3w, D[0].ln3b, SC, ntok);
    // layer 1
    sa_block(SC, saWtD[1], D[1].sab, saWotD[1], D[1].sabo, D[1].ln1w, D[1].ln1b, SA, ntok);
    ca_block(SA, 1, SB, ntok);
    ffn_block(SB, W1tD[1], D[1].b1, W2tD[1], D[1].b2, D[1].ln3w, D[1].ln3b, SD, ntok);
    commit_kernel<<<NB, 64, 0, stream>>>(SD, BSTR, BUFX, BSTR, (float*)d_out, t);
  }
}
